// HGT_20289425506750
// MI455X (gfx1250) — matrix operand order repaired and checked by reading, not compiled, not run on hardware
//
#include <hip/hip_runtime.h>
#include <hip/hip_bf16.h>
#include <math.h>

typedef __attribute__((ext_vector_type(16))) _Float16 v16h;
typedef __attribute__((ext_vector_type(8)))  _Float16 v8h;
typedef __attribute__((ext_vector_type(8)))  float    v8f;
typedef __attribute__((ext_vector_type(4)))  unsigned v4u;
typedef __attribute__((ext_vector_type(8)))  int      v8i;
typedef __attribute__((ext_vector_type(4)))  int      v4i;

// LDS (address_space(3)) typed views so laundered pointers still lower to
// ds_load_b128 instead of flat_load_b128.
typedef __attribute__((address_space(3))) _Float16 lds_h;
typedef __attribute__((address_space(3))) v8h      lds_v8h;

#define NP_    100000
#define NA_    50000
#define CC     128
#define HH     8
#define DD     16
#define LL     2
#define GG     64
#define NOUT   64
#define NEDGE  500000

// Use the Tensor Data Mover when the builtin is available in the device pass;
// this toolchain exposes the 6-arg (clang-23 / therock-10.0) form:
//   (uint32x4 g0, int32x8 g1, int32x4 g2, int32x4 g3, int32x8 g4, i32 cpol)
#if defined(__HIP_DEVICE_COMPILE__)
#if __has_builtin(__builtin_amdgcn_tensor_load_to_lds)
#define USE_TDM 1
#endif
#endif
#ifndef USE_TDM
#define USE_TDM 0
#endif

// ---------------------------------------------------------------------------
// order-preserving float <-> uint encoding for atomicMax-based segment max
// ---------------------------------------------------------------------------
__device__ __forceinline__ unsigned enc_f32(float x) {
    unsigned u = __float_as_uint(x);
    return (u & 0x80000000u) ? ~u : (u | 0x80000000u);
}
__device__ __forceinline__ float dec_f32(unsigned k) {
    return (k & 0x80000000u) ? __uint_as_float(k & 0x7FFFFFFFu)
                             : __uint_as_float(~k);
}
#define ENC_NEG_INF 0x007FFFFFu  // enc_f32(-inf)

// ---------------------------------------------------------------------------
// weight staging: f32 [in][out] -> f16 [out][in]  (B fragments become
// two contiguous 8-half runs along K)
// ---------------------------------------------------------------------------
__global__ void k_pack_plain(const float* __restrict__ W, _Float16* __restrict__ Wt) {
    int t = blockIdx.x * blockDim.x + threadIdx.x;
    if (t >= CC * CC) return;
    int o = t >> 7, i = t & 127;
    Wt[o * CC + i] = (_Float16)W[i * CC + o];
}

// composite weight: Weff[i][h*16+eo] = sum_d W[i][h*16+d] * rel[h][d][eo]
// (folds the per-head a_rel/m_rel einsum into the projection), stored
// transposed f16; also folds the bias.
__global__ void k_pack_rel(const float* __restrict__ W, const float* __restrict__ b,
                           const float* __restrict__ rel,
                           _Float16* __restrict__ Wt, float* __restrict__ beff) {
    int t = blockIdx.x * blockDim.x + threadIdx.x;
    if (t >= CC * CC) return;
    int o = t >> 7, i = t & 127;
    int h = o >> 4, eo = o & 15;
    const float* Wc = W + i * CC + h * DD;
    const float* R  = rel + h * DD * DD + eo;
    float acc = 0.f;
#pragma unroll
    for (int d = 0; d < DD; ++d) acc += Wc[d] * R[d * DD];
    Wt[o * CC + i] = (_Float16)acc;
    if (i == 0) {
        const float* bc = b + h * DD;
        float bb = 0.f;
#pragma unroll
        for (int d = 0; d < DD; ++d) bb += bc[d] * R[d * DD];
        beff[o] = bb;
    }
}

__global__ void k_to_half(const float* __restrict__ in, _Float16* __restrict__ out, int n) {
    int t = blockIdx.x * blockDim.x + threadIdx.x;
    if (t < n) out[t] = (_Float16)in[t];
}

__global__ void k_gelu_to_half(const float* __restrict__ in, _Float16* __restrict__ out, int n) {
    int t = blockIdx.x * blockDim.x + threadIdx.x;
    if (t < n) {
        float x = in[t];
        float g = 0.5f * x * (1.0f + erff(x * 0.70710678118654752f));
        out[t] = (_Float16)g;
    }
}

__global__ void k_fill(unsigned* __restrict__ p, unsigned v, int n) {
    int t = blockIdx.x * blockDim.x + threadIdx.x;
    if (t < n) p[t] = v;
}

// ---------------------------------------------------------------------------
// WMMA GEMM:  C[N x 128] = A[N x 128] * Bt^T + bias, optional epilogue.
// B (128x128 f16, 32 KB) is staged into LDS once per workgroup by the Tensor
// Data Mover (TENSORcnt-tracked), then all 8 waves read B fragments with
// ds_load_b128. A streams from global with b128 loads.
// mode 0: none   mode 1: relu   mode 2: beta*v + (1-beta)*skipx
// ---------------------------------------------------------------------------
__global__ __launch_bounds__(256)
void k_gemm(const _Float16* __restrict__ A, const _Float16* __restrict__ Bt,
            const float* __restrict__ bias, float* __restrict__ Cf,
            _Float16* __restrict__ Ch, const float* __restrict__ skipx,
            const float* __restrict__ skipparam, int N, int mode) {
    __shared__ __align__(16) _Float16 Bs[CC * CC];

    // Launder the 32-bit LDS byte offset of Bs (not the generic pointer!):
    // - opaque value => the optimizer cannot prove Bs is never stored, so the
    //   B-fragment loads are not folded to undef (wmma collapsed 32->4 before)
    // - rebuilt as an address_space(3) pointer => backend emits ds_load_b128
    //   (laundering the generic pointer instead produced flat_load_b128)
    unsigned bso = (unsigned)(size_t)&Bs[0];   // flat addr[31:0] = LDS offset
    asm volatile("" : "+v"(bso));
    lds_h* bsp = (lds_h*)bso;

#if USE_TDM
    if (threadIdx.x < 32) {
        // ---- Tensor DMA descriptor (D#) per CDNA5 ISA §8 ----
        unsigned long long ga = (unsigned long long)(size_t)Bt;
        v4u g0;
        g0[0] = 1u;                                   // count=1, user mode
        g0[1] = bso;                                  // lds_addr (bytes)
        g0[2] = (unsigned)(ga & 0xFFFFFFFFull);       // global_addr[31:0]
        g0[3] = (unsigned)((ga >> 32) & 0x01FFFFFFull) | (2u << 30); // [56:32]|type=2
        v8i g1;
        g1[0] = (int)(1u << 16);            // wg_mask=0, data_size=1 (2 bytes)
        g1[1] = (int)(CC << 16);            // tensor_dim0[15:0]  (bits 63:48)
        g1[2] = (int)(CC << 16);            // dim0 hi=0 | tensor_dim1[15:0]
        g1[3] = (int)(CC << 16);            // dim1 hi=0 | tile_dim0=128
        g1[4] = (int)CC;                    // tile_dim1=128, tile_dim2=0
        g1[5] = (int)CC;                    // tensor_dim0_stride[31:0]=128
        g1[6] = (int)((CC * CC) << 16);     // stride0 hi=0 | dim1_stride lo=16384
        g1[7] = 0;                          // dim1_stride hi
        v4i gz4 = {0, 0, 0, 0};             // groups 2/3 unused (2-D tensor)
        v8i gz8 = {0, 0, 0, 0, 0, 0, 0, 0};
        __builtin_amdgcn_tensor_load_to_lds(g0, g1, gz4, gz4, gz8, 0);
        __builtin_amdgcn_s_wait_tensorcnt(0);
    }
    asm volatile("" : : : "memory");
#else
    for (int i = threadIdx.x; i < (CC * CC) / 8; i += blockDim.x)
        ((lds_v8h*)bsp)[i] = ((const v8h*)Bt)[i];
#endif
    __syncthreads();

    int wave = threadIdx.x >> 5;
    int lane = threadIdx.x & 31;
    int base = blockIdx.x * 128 + wave * 16;
    if (base >= N) return;

    int hi16 = (lane < 16) ? 0 : 1;
    int l15  = lane & 15;

    v8f zero = {};
    v8f acc[8];
#pragma unroll
    for (int ct = 0; ct < 8; ++ct) acc[ct] = zero;

    const _Float16* Arow = A + (size_t)(base + l15) * CC;
    int kbA = hi16 * 8;

#pragma unroll
    for (int kk = 0; kk < 4; ++kk) {
        int k0 = kk * 32;
        union { v16h v; v8h h[2]; } af;
        af.h[0] = *(const v8h*)(Arow + k0 + kbA);
        af.h[1] = *(const v8h*)(Arow + k0 + kbA + 16);
        int koffB = k0 + hi16 * 8;
#pragma unroll
        for (int ct = 0; ct < 8; ++ct) {
            union { v16h v; v8h h[2]; } bf;
            lds_h* Bp = bsp + (ct * 16 + l15) * CC + koffB;
            bf.h[0] = *(const lds_v8h*)(Bp);
            bf.h[1] = *(const lds_v8h*)(Bp + 16);
            acc[ct] = __builtin_amdgcn_wmma_f32_16x16x32_f16(
                false, af.v, false, bf.v, (short)0, acc[ct], false, false);
        }
    }

    float beta = 0.f, ombeta = 0.f;
    if (mode == 2) {
        float s = skipparam[0];
        beta   = 1.f / (1.f + __expf(-s));
        ombeta = 1.f - beta;
    }

#pragma unroll
    for (int ct = 0; ct < 8; ++ct) {
        int col = ct * 16 + l15;
        float bv = bias[col];
#pragma unroll
        for (int r = 0; r < 8; ++r) {
            int row = base + r + hi16 * 8;
            size_t idx = (size_t)row * CC + col;
            float v = acc[ct][r] + bv;
            if (mode == 1)      v = fmaxf(v, 0.f);
            else if (mode == 2) v = beta * v + ombeta * skipx[idx];
            if (Cf) Cf[idx] = v;
            if (Ch) Ch[idx] = (_Float16)v;
        }
    }
}

// ---------------------------------------------------------------------------
// edge phase: 3-pass atomic segment softmax + message scatter
// ---------------------------------------------------------------------------
__global__ void k_alpha(const float* __restrict__ q, const float* __restrict__ krel,
                        const int* __restrict__ src, const int* __restrict__ dst,
                        const float* __restrict__ prel,
                        float* __restrict__ alpha, unsigned* __restrict__ amax) {
    int t = blockIdx.x * blockDim.x + threadIdx.x;
    if (t >= NEDGE * HH) return;
    int e = t >> 3, h = t & 7;
    int s = src[e], d = dst[e];
    const float4* qp = (const float4*)(q    + (size_t)d * CC + h * DD);
    const float4* kp = (const float4*)(krel + (size_t)s * CC + h * DD);
    float acc = 0.f;
#pragma unroll
    for (int j = 0; j < 4; ++j) {
        float4 qv = qp[j], kv = kp[j];
        acc += qv.x * kv.x + qv.y * kv.y + qv.z * kv.z + qv.w * kv.w;
    }
    float a = acc * prel[h] * 0.25f;  // 1/sqrt(16)
    alpha[t] = a;
    atomicMax(&amax[(size_t)d * HH + h], enc_f32(a));
}

__global__ void k_expsum(float* __restrict__ alpha, const int* __restrict__ dst,
                         const unsigned* __restrict__ amax, float* __restrict__ asum) {
    int t = blockIdx.x * blockDim.x + threadIdx.x;
    if (t >= NEDGE * HH) return;
    int e = t >> 3, h = t & 7;
    int d = dst[e];
    float m  = dec_f32(amax[(size_t)d * HH + h]);
    float ex = __expf(alpha[t] - m);
    alpha[t] = ex;
    atomicAdd(&asum[(size_t)d * HH + h], ex);
}

// one wave per edge, each lane owns 4 contiguous channels (single head/lane)
__global__ void k_scatter(const float* __restrict__ alpha, const float* __restrict__ asum,
                          const float* __restrict__ vrel, const int* __restrict__ src,
                          const int* __restrict__ dst, float* __restrict__ out) {
    int e = blockIdx.x * (blockDim.x >> 5) + (threadIdx.x >> 5);
    if (e >= NEDGE) return;
    int lane = threadIdx.x & 31;
    int c0 = lane * 4;
    int h  = c0 >> 4;
    int s = src[e], d = dst[e];
    float att = alpha[(size_t)e * HH + h] / asum[(size_t)d * HH + h];
    const float4 v = *(const float4*)(vrel + (size_t)s * CC + c0);
    float* op = out + (size_t)d * CC + c0;
    atomicAdd(op + 0, v.x * att);
    atomicAdd(op + 1, v.y * att);
    atomicAdd(op + 2, v.z * att);
    atomicAdd(op + 3, v.w * att);
}

// ---------------------------------------------------------------------------
// pooling + final projection
// ---------------------------------------------------------------------------
__global__ void k_pool_add(const float* __restrict__ x, const int* __restrict__ batch,
                           float* __restrict__ hsum, int N) {
    int t = blockIdx.x * blockDim.x + threadIdx.x;
    if (t >= N * CC) return;
    int n = t >> 7, c = t & 127;
    atomicAdd(&hsum[(size_t)batch[n] * CC + c], x[t]);
}

__global__ void k_pool_cnt(const int* __restrict__ batch, float* __restrict__ cnt, int N) {
    int t = blockIdx.x * blockDim.x + threadIdx.x;
    if (t < N) atomicAdd(&cnt[batch[t]], 1.f);
}

__global__ void k_final(const float* __restrict__ hsum_p, const float* __restrict__ cnt_p,
                        const float* __restrict__ hsum_a, const float* __restrict__ cnt_a,
                        const float* __restrict__ Wout, const float* __restrict__ bout,
                        float* __restrict__ out) {
    int t = blockIdx.x * blockDim.x + threadIdx.x;
    if (t >= GG * NOUT) return;
    int g = t / NOUT, o = t % NOUT;
    float ip = 1.f / fmaxf(cnt_p[g], 1.f);
    float ia = 1.f / fmaxf(cnt_a[g], 1.f);
    float acc = 0.f;
    for (int c = 0; c < CC; ++c) {
        float hg = hsum_p[g * CC + c] * ip + hsum_a[g * CC + c] * ia;
        acc += hg * Wout[c * NOUT + o];
    }
    out[t] = acc + bout[o];
}

// ---------------------------------------------------------------------------
extern "C" void kernel_launch(void* const* d_in, const int* in_sizes, int n_in,
                              void* d_out, int out_size, void* d_ws, size_t ws_size,
                              hipStream_t stream) {
    (void)in_sizes; (void)n_in; (void)out_size; (void)ws_size;
    const float* x_p    = (const float*)d_in[0];
    const float* x_a    = (const float*)d_in[1];
    const float* Wlin   = (const float*)d_in[2];
    const float* blin   = (const float*)d_in[3];
    const float* Wk     = (const float*)d_in[4];
    const float* bk     = (const float*)d_in[5];
    const float* Wq     = (const float*)d_in[6];
    const float* bq     = (const float*)d_in[7];
    const float* Wv     = (const float*)d_in[8];
    const float* bv     = (const float*)d_in[9];
    const float* a_rel  = (const float*)d_in[10];
    const float* m_rel  = (const float*)d_in[11];
    const float* p_rel  = (const float*)d_in[12];
    const float* Wa     = (const float*)d_in[13];
    const float* ba     = (const float*)d_in[14];
    const float* skip   = (const float*)d_in[15];
    const float* Wout   = (const float*)d_in[16];
    const float* bout   = (const float*)d_in[17];
    const int* e_src[3] = {(const int*)d_in[18], (const int*)d_in[20], (const int*)d_in[22]};
    const int* e_dst[3] = {(const int*)d_in[19], (const int*)d_in[21], (const int*)d_in[23]};
    const int* batch_p  = (const int*)d_in[24];
    const int* batch_a  = (const int*)d_in[25];
    float* out = (float*)d_out;

    // ---- workspace carve-out ----
    char* ws = (char*)d_ws;
    size_t off = 0;
    auto alloc = [&](size_t bytes) -> char* {
        char* p = ws + off;
        off += (bytes + 255) & ~(size_t)255;
        return p;
    };
    const int Ns[2] = {NP_, NA_};
    float*    xs[2];  _Float16* xs16[2];  float* q[2];  float* outb[2];  _Float16* g16[2];
    for (int t = 0; t < 2; ++t) {
        xs[t]   = (float*)alloc((size_t)Ns[t] * CC * 4);
        q[t]    = (float*)alloc((size_t)Ns[t] * CC * 4);
        outb[t] = (float*)alloc((size_t)Ns[t] * CC * 4);
        xs16[t] = (_Float16*)alloc((size_t)Ns[t] * CC * 2);
        g16[t]  = (_Float16*)alloc((size_t)Ns[t] * CC * 2);
    }
    float*    krel  = (float*)alloc((size_t)NP_ * CC * 4);
    float*    vrel  = (float*)alloc((size_t)NP_ * CC * 4);
    float*    alpha = (float*)alloc((size_t)NEDGE * HH * 4);
    unsigned* amax  = (unsigned*)alloc((size_t)NP_ * HH * 4);
    float*    asum  = (float*)alloc((size_t)NP_ * HH * 4);
    _Float16* WtLin[2]; _Float16* WtQ[LL][2]; _Float16* WtA[LL][2];
    _Float16* WtK[LL][3]; _Float16* WtV[LL][3];
    float*    bK[LL][3];  float*    bV[LL][3];
    for (int t = 0; t < 2; ++t) WtLin[t] = (_Float16*)alloc(CC * CC * 2);
    for (int l = 0; l < LL; ++l)
        for (int t = 0; t < 2; ++t) {
            WtQ[l][t] = (_Float16*)alloc(CC * CC * 2);
            WtA[l][t] = (_Float16*)alloc(CC * CC * 2);
        }
    for (int l = 0; l < LL; ++l)
        for (int e = 0; e < 3; ++e) {
            WtK[l][e] = (_Float16*)alloc(CC * CC * 2);
            WtV[l][e] = (_Float16*)alloc(CC * CC * 2);
            bK[l][e]  = (float*)alloc(CC * 4);
            bV[l][e]  = (float*)alloc(CC * 4);
        }
    float* hsum_p = (float*)alloc(GG * CC * 4);
    float* hsum_a = (float*)alloc(GG * CC * 4);
    float* cnt_p  = (float*)alloc(GG * 4);
    float* cnt_a  = (float*)alloc(GG * 4);

    const int st_of[3] = {0, 1, 0};
    const int dt_of[3] = {0, 0, 1};
    dim3 B(256);
    int gW = (CC * CC + 255) / 256;

    // ---- stage all weights (fold a_rel/m_rel into K/V projections) ----
    for (int t = 0; t < 2; ++t)
        k_pack_plain<<<gW, B, 0, stream>>>(Wlin + t * CC * CC, WtLin[t]);
    for (int l = 0; l < LL; ++l) {
        for (int t = 0; t < 2; ++t) {
            k_pack_plain<<<gW, B, 0, stream>>>(Wq + (l * 2 + t) * CC * CC, WtQ[l][t]);
            k_pack_plain<<<gW, B, 0, stream>>>(Wa + (l * 2 + t) * CC * CC, WtA[l][t]);
        }
        for (int e = 0; e < 3; ++e) {
            int st = st_of[e];
            k_pack_rel<<<gW, B, 0, stream>>>(Wk + (l * 2 + st) * CC * CC,
                                             bk + (l * 2 + st) * CC,
                                             a_rel + (l * 3 + e) * HH * DD * DD,
                                             WtK[l][e], bK[l][e]);
            k_pack_rel<<<gW, B, 0, stream>>>(Wv + (l * 2 + st) * CC * CC,
                                             bv + (l * 2 + st) * CC,
                                             m_rel + (l * 3 + e) * HH * DD * DD,
                                             WtV[l][e], bV[l][e]);
        }
    }

    // ---- input projection + relu (writes f32 xs and f16 shadow) ----
    for (int t = 0; t < 2; ++t) {
        int nel = Ns[t] * CC;
        k_to_half<<<(nel + 255) / 256, B, 0, stream>>>(t == 0 ? x_p : x_a, g16[t], nel);
        k_gemm<<<(Ns[t] + 127) / 128, B, 0, stream>>>(
            g16[t], WtLin[t], blin + t * CC, xs[t], xs16[t],
            nullptr, nullptr, Ns[t], /*relu*/ 1);
    }

    // ---- layers ----
    for (int l = 0; l < LL; ++l) {
        for (int t = 0; t < 2; ++t)
            k_gemm<<<(Ns[t] + 127) / 128, B, 0, stream>>>(
                xs16[t], WtQ[l][t], bq + (l * 2 + t) * CC, q[t], nullptr,
                nullptr, nullptr, Ns[t], 0);
        for (int t = 0; t < 2; ++t)
            k_fill<<<(Ns[t] * CC + 255) / 256, B, 0, stream>>>(
                (unsigned*)outb[t], 0u, Ns[t] * CC);

        for (int e = 0; e < 3; ++e) {
            int st = st_of[e], dt = dt_of[e];
            k_gemm<<<(Ns[st] + 127) / 128, B, 0, stream>>>(
                xs16[st], WtK[l][e], bK[l][e], krel, nullptr, nullptr, nullptr, Ns[st], 0);
            k_gemm<<<(Ns[st] + 127) / 128, B, 0, stream>>>(
                xs16[st], WtV[l][e], bV[l][e], vrel, nullptr, nullptr, nullptr, Ns[st], 0);
            int nseg = Ns[dt] * HH;
            k_fill<<<(nseg + 255) / 256, B, 0, stream>>>(amax, ENC_NEG_INF, nseg);
            k_fill<<<(nseg + 255) / 256, B, 0, stream>>>((unsigned*)asum, 0u, nseg);
            int gE = (NEDGE * HH + 255) / 256;
            k_alpha<<<gE, B, 0, stream>>>(q[dt], krel, e_src[e], e_dst[e],
                                          p_rel + (l * 3 + e) * HH, alpha, amax);
            k_expsum<<<gE, B, 0, stream>>>(alpha, e_dst[e], amax, asum);
            k_scatter<<<(NEDGE + 7) / 8, B, 0, stream>>>(alpha, asum, vrel,
                                                         e_src[e], e_dst[e], outb[dt]);
        }

        for (int t = 0; t < 2; ++t) {
            int nel = Ns[t] * CC;
            k_gelu_to_half<<<(nel + 255) / 256, B, 0, stream>>>(outb[t], g16[t], nel);
            k_gemm<<<(Ns[t] + 127) / 128, B, 0, stream>>>(
                g16[t], WtA[l][t], ba + (l * 2 + t) * CC, xs[t], xs16[t],
                xs[t], skip + (l * 2 + t), Ns[t], /*gated skip*/ 2);
        }
    }

    // ---- pooling + final projection ----
    k_fill<<<(GG * CC + 255) / 256, B, 0, stream>>>((unsigned*)hsum_p, 0u, GG * CC);
    k_fill<<<(GG * CC + 255) / 256, B, 0, stream>>>((unsigned*)hsum_a, 0u, GG * CC);
    k_fill<<<1, B, 0, stream>>>((unsigned*)cnt_p, 0u, GG);
    k_fill<<<1, B, 0, stream>>>((unsigned*)cnt_a, 0u, GG);
    k_pool_add<<<(NP_ * CC + 255) / 256, B, 0, stream>>>(xs[0], batch_p, hsum_p, NP_);
    k_pool_add<<<(NA_ * CC + 255) / 256, B, 0, stream>>>(xs[1], batch_a, hsum_a, NA_);
    k_pool_cnt<<<(NP_ + 255) / 256, B, 0, stream>>>(batch_p, cnt_p, NP_);
    k_pool_cnt<<<(NA_ + 255) / 256, B, 0, stream>>>(batch_a, cnt_a, NA_);
    k_final<<<(GG * NOUT + 255) / 256, B, 0, stream>>>(hsum_p, cnt_p, hsum_a, cnt_a,
                                                       Wout, bout, out);
}
